// Bn_bin_conv_pool_block_baw_27101243638455
// MI455X (gfx1250) — 8-bit matrix operand order repaired and checked by reading, compiled, not run on hardware
//
#include <hip/hip_runtime.h>
#include <hip/hip_bf16.h>
#include <stdint.h>

typedef __attribute__((ext_vector_type(8))) int v8i;

#define CIN        64
#define COUT       128
#define KTAP       7
#define LIN        4096
#define LOUT       4096
#define LPOOL      2048
#define BATCH      256
#define LTILE      128
#define LHALO      (LTILE + KTAP - 1)
#define WPACK_BYTES (KTAP * COUT * CIN)
#define STATS_OFF   65536

__global__ void pack_weights_kernel(const float* __restrict__ W,
                                    int8_t* __restrict__ wp) {
    int tid = blockIdx.x * blockDim.x + threadIdx.x;
    if (tid >= WPACK_BYTES) return;
    int k  = tid / (COUT * CIN);
    int r  = tid % (COUT * CIN);
    int co = r / CIN;
    int ci = r % CIN;
    float v = W[(co * CIN + ci) * KTAP + k];
    wp[tid] = (v > 0.f) ? (int8_t)1 : ((v < 0.f) ? (int8_t)-1 : (int8_t)0);
}

__global__ __launch_bounds__(256)
void bconv_pool_prelu_kernel(const float* __restrict__ x,
                             const int8_t* __restrict__ wp,
                             const float* __restrict__ alphap,
                             float* __restrict__ out) {
    __shared__ __align__(16) int8_t ldsW[KTAP * 64 * CIN];
    __shared__ __align__(16) int8_t ldsA[LHALO * CIN];

    const int tid   = threadIdx.x;
    const int lane  = tid & 31;
    const int wave  = tid >> 5;
    const int col   = lane & 15;
    const int half  = lane >> 4;
    const int ltile = blockIdx.x;
    const int chalf = blockIdx.y;
    const int b     = blockIdx.z;
    const int lbase = ltile * LTILE;

    const float alpha = alphap[0];

    {
        const uint4* src4 = (const uint4*)(wp);
        uint4* dst4 = (uint4*)ldsW;
        #pragma unroll
        for (int it = 0; it < 7; ++it) {
            int i = tid + it * 256;
            int k = i >> 8;
            int r = i & 255;
            dst4[i] = src4[(k * 8192 + chalf * 4096) / 16 + r];
        }
    }

    {
        const int nitems = LHALO * (CIN / 4);
        for (int i = tid; i < nitems; i += 256) {
            int l   = i >> 4;
            int cig = i & 15;
            int pos = lbase - 3 + l;
            uint32_t pk = 0;
            #pragma unroll
            for (int j = 0; j < 4; ++j) {
                int ci = cig * 4 + j;
                float v = (pos >= 0 && pos < LIN)
                            ? x[((size_t)b * CIN + ci) * LIN + pos]
                            : -1.0f;
                int s = (v > 0.f) ? 1 : ((v < 0.f) ? -1 : 0);
                pk |= (uint32_t)(s & 0xFF) << (8 * j);
            }
            *(uint32_t*)(ldsA + l * CIN + cig * 4) = pk;
        }
    }
    __syncthreads();

    const int co_tile = wave >> 1;
    const int nbase   = (wave & 1) * 4;

    const int8_t* wbase = ldsW + ((co_tile * 16 + col) * CIN) + half * 16;
    const int8_t* brow0 = ldsA + (((nbase + 0) * 16 + col) * CIN) + half * 16;
    const int8_t* brow1 = ldsA + (((nbase + 1) * 16 + col) * CIN) + half * 16;
    const int8_t* brow2 = ldsA + (((nbase + 2) * 16 + col) * CIN) + half * 16;
    const int8_t* brow3 = ldsA + (((nbase + 3) * 16 + col) * CIN) + half * 16;

    auto loadA = [&](int k) -> v8i {
        const int8_t* p = wbase + k * (64 * CIN);
        int2 t0 = *(const int2*)(p);
        int2 t1 = *(const int2*)(p + 8);
        int2 t2 = *(const int2*)(p + 32);
        int2 t3 = *(const int2*)(p + 40);
        return (v8i){t0.x, t0.y, t1.x, t1.y, t2.x, t2.y, t3.x, t3.y};
    };
    auto loadB = [&](const int8_t* brow, int k) -> v8i {
        const int8_t* p = brow + k * CIN;
        int4 q0 = *(const int4*)(p);
        int4 q1 = *(const int4*)(p + 32);
        return (v8i){q0.x, q0.y, q0.z, q0.w, q1.x, q1.y, q1.z, q1.w};
    };

    v8i acc0 = (v8i){0,0,0,0,0,0,0,0};
    v8i acc1 = acc0, acc2 = acc0, acc3 = acc0;

    v8i aCur = loadA(0);
    v8i bC0 = loadB(brow0, 0);
    v8i bC1 = loadB(brow1, 0);
    v8i bC2 = loadB(brow2, 0);
    v8i bC3 = loadB(brow3, 0);

    #pragma unroll
    for (int k = 0; k < KTAP; ++k) {
        v8i aNext, bN0, bN1, bN2, bN3;
        if (k < KTAP - 1) {
            aNext = loadA(k + 1);
            bN0 = loadB(brow0, k + 1);
            bN1 = loadB(brow1, k + 1);
            bN2 = loadB(brow2, k + 1);
            bN3 = loadB(brow3, k + 1);
        }
        acc0 = __builtin_amdgcn_wmma_i32_16x16x64_iu8(true, aCur, true, bC0,
                                                      acc0, false, false);
        acc1 = __builtin_amdgcn_wmma_i32_16x16x64_iu8(true, aCur, true, bC1,
                                                      acc1, false, false);
        acc2 = __builtin_amdgcn_wmma_i32_16x16x64_iu8(true, aCur, true, bC2,
                                                      acc2, false, false);
        acc3 = __builtin_amdgcn_wmma_i32_16x16x64_iu8(true, aCur, true, bC3,
                                                      acc3, false, false);
        if (k < KTAP - 1) {
            aCur = aNext;
            bC0 = bN0; bC1 = bN1; bC2 = bN2; bC3 = bN3;
        }
    }

    v8i accs[4] = {acc0, acc1, acc2, acc3};
    #pragma unroll
    for (int n = 0; n < 4; ++n) {
        const int lout = lbase + (nbase + n) * 16 + col;
        #pragma unroll
        for (int r = 0; r < 8; ++r) {
            int v  = accs[n][r];
            int o  = __builtin_amdgcn_ds_swizzle(v, 0x041F);
            int m  = (v > o) ? v : o;
            float f = (float)m;
            f = (f > 0.f) ? f : alpha * f;
            if ((col & 1) == 0) {
                int co = chalf * 64 + co_tile * 16 + r + half * 8;
                int lp = lout >> 1;
                out[((size_t)b * COUT + co) * LPOOL + lp] = f;
            }
        }
    }
}

__global__ __launch_bounds__(256)
void stats_kernel(const float* __restrict__ y, float* __restrict__ stats) {
    __shared__ float ssum[256];
    __shared__ float ssq[256];
    const int c   = blockIdx.x;
    const int tid = threadIdx.x;
    const int N   = BATCH * LPOOL;

    float s = 0.f, q = 0.f;
    for (int i = tid; i < N; i += 256) {
        int bb = i / LPOOL;
        int lp = i % LPOOL;
        float v = y[((size_t)bb * COUT + c) * LPOOL + lp];
        s += v;
        q += v * v;
    }
    ssum[tid] = s;
    ssq[tid]  = q;
    __syncthreads();
    for (int off = 128; off > 0; off >>= 1) {
        if (tid < off) {
            ssum[tid] += ssum[tid + off];
            ssq[tid]  += ssq[tid + off];
        }
        __syncthreads();
    }
    if (tid == 0) {
        float mean = ssum[0] / (float)N;
        float var  = ssq[0] / (float)N - mean * mean;
        stats[c]        = mean;
        stats[COUT + c] = __frsqrt_rn(var + 1e-5f);
    }
}

__global__ __launch_bounds__(256)
void bn_apply_kernel(float* __restrict__ y,
                     const float* __restrict__ stats,
                     const float* __restrict__ gamma,
                     const float* __restrict__ beta) {
    size_t i4 = (size_t)blockIdx.x * blockDim.x + threadIdx.x;
    size_t base = i4 * 4;
    int c = (int)((base >> 11) & (COUT - 1));
    float mean = stats[c];
    float sc   = stats[COUT + c] * gamma[c];
    float bi   = beta[c] - mean * sc;
    float4 v = ((float4*)y)[i4];
    v.x = v.x * sc + bi;
    v.y = v.y * sc + bi;
    v.z = v.z * sc + bi;
    v.w = v.w * sc + bi;
    ((float4*)y)[i4] = v;
}

extern "C" void kernel_launch(void* const* d_in, const int* in_sizes, int n_in,
                              void* d_out, int out_size, void* d_ws, size_t ws_size,
                              hipStream_t stream) {
    const float* x     = (const float*)d_in[0];
    const float* W     = (const float*)d_in[1];
    const float* alpha = (const float*)d_in[2];
    const float* gamma = (const float*)d_in[3];
    const float* beta  = (const float*)d_in[4];
    float* out = (float*)d_out;

    int8_t* wp   = (int8_t*)d_ws;
    float* stats = (float*)((char*)d_ws + STATS_OFF);

    pack_weights_kernel<<<(WPACK_BYTES + 255) / 256, 256, 0, stream>>>(W, wp);

    dim3 grid(LOUT / LTILE, 2, BATCH);
    bconv_pool_prelu_kernel<<<grid, 256, 0, stream>>>(x, wp, alpha, out);

    stats_kernel<<<COUT, 256, 0, stream>>>(out, stats);

    size_t n4 = (size_t)out_size / 4;
    bn_apply_kernel<<<(unsigned)(n4 / 256), 256, 0, stream>>>(out, stats, gamma, beta);
}
